// ScaledNeuromorphicBrain_1425929142380
// MI455X (gfx1250) — matrix operand order repaired and checked by reading, not compiled, not run on hardware
//
#include <hip/hip_runtime.h>

// ---------------------------------------------------------------------------
// CDNA5 / gfx1250 bf16 WMMA transformer forward
// ---------------------------------------------------------------------------

typedef __attribute__((ext_vector_type(16))) __bf16 v16bf;
typedef __attribute__((ext_vector_type(8)))  float  v8f;

#define BM 128
#define BN 128
#define BKK 32
#define LDAS 40  // A row stride (elems): 80B rows -> 16B aligned, 20-bank step
#define LDBS 40  // B column stride (elems): sB is column-major [n][k]

// ---------------------------------------------------------------------------
// Generic GEMM: C[M,N] = A[M,K](bf16) x B[K,N](f32, converted) (+bias)(+resid)
// BT=1: B is stored [N,K] row-major (used for logits = x @ emb^T).
// Optional outputs: C (f32), Cbf (bf16 row-major), Ct (bf16 transposed
//   [row>>10][col][row&1023] -> per-batch column-major, for attention V^T).
// A tile is staged via GLOBAL_LOAD_ASYNC_TO_LDS_B128 (ASYNCcnt path);
// B tile needs f32->bf16 conversion so it is staged with VALU converts.
// ---------------------------------------------------------------------------
template <bool BT>
__global__ __launch_bounds__(256) void gemm_kernel(
    const __bf16* __restrict__ A, const float* __restrict__ Bm,
    const float* __restrict__ bias, const float* __restrict__ resid,
    float* __restrict__ C, __bf16* __restrict__ Cbf, __bf16* __restrict__ Ct,
    int M, int N, int K) {
  __shared__ __bf16 sA[BM * LDAS];  // 10 KB
  __shared__ __bf16 sB[BN * LDBS];  // 10 KB, column-major [n][k]

  const int tid  = threadIdx.x;
  const int lane = tid & 31;
  const int wave = tid >> 5;
  const int wm   = wave >> 1;  // 0..3 -> 32-row strip
  const int wn   = wave & 1;   // 0..1 -> 64-col strip
  const int n0   = blockIdx.x * BN;
  const int m0   = blockIdx.y * BM;

  v8f acc[2][4];
#pragma unroll
  for (int i = 0; i < 2; i++)
#pragma unroll
    for (int j = 0; j < 4; j++) acc[i][j] = {};

  for (int kb = 0; kb < K; kb += BKK) {
    // ---- stage A tile: async DMA bf16 global -> LDS (2 threads/row, 32B ea)
    {
      const int r  = tid >> 1;
      const int c0 = (tid & 1) * 16;
      __bf16* dst = sA + r * LDAS + c0;
      if (m0 + r < M) {
        const __bf16* src = A + (size_t)(m0 + r) * K + kb + c0;
        const unsigned dstoff = (unsigned)(uintptr_t)dst;  // LDS byte address
        asm volatile(
            "global_load_async_to_lds_b128 %0, %1, off\n\t"
            "global_load_async_to_lds_b128 %0, %1, off offset:16"
            :
            : "v"(dstoff), "v"(src)
            : "memory");
      } else {
#pragma unroll
        for (int i = 0; i < 16; i++) dst[i] = (__bf16)0.f;
      }
    }
    // ---- stage B tile (f32 global -> bf16 LDS, column-major)
    if (!BT) {
      const int n  = tid & 127;        // column within tile
      const int kh = (tid >> 7) * 16;  // 0 or 16
      const bool ok = (n0 + n) < N;
      __bf16* dst = sB + n * LDBS + kh;
#pragma unroll
      for (int i = 0; i < 16; i++) {
        const float v = ok ? Bm[(size_t)(kb + kh + i) * N + n0 + n] : 0.f;
        dst[i] = (__bf16)v;  // lanes walk n -> coalesced global loads
      }
    } else {
      const int n  = tid >> 1;        // 0..127
      const int k0 = (tid & 1) * 16;  // 0 or 16
      const float* src = Bm + (size_t)(n0 + n) * K + kb + k0;
      const bool ok = (n0 + n) < N;
      __bf16* dst = sB + n * LDBS + k0;
#pragma unroll
      for (int i = 0; i < 16; i++) dst[i] = ok ? (__bf16)src[i] : (__bf16)0.f;
    }
    asm volatile("s_wait_asynccnt 0x0" ::: "memory");
    __syncthreads();

    // prefetch next A tile (gfx1250 global_prefetch_b8 path)
    if (kb + BKK < K && m0 + (tid >> 1) < M)
      __builtin_prefetch(A + (size_t)(m0 + (tid >> 1)) * K + kb + BKK, 0, 0);

    // ---- WMMA: per-wave 2 m-tiles x 4 n-tiles, one 16x16x32 step
    const int mrow  = lane & 15;
    const int koff  = (lane >> 4) * 16;  // A: K 0..15 vs 16..31 halves
    const int ncol  = lane & 15;
    const int kbase = (lane >> 4) * 16;  // B: K 0..15 vs 16..31 halves
    v16bf afrag[2];
#pragma unroll
    for (int mt = 0; mt < 2; mt++) {
      const int row = (wm * 32 + mt * 16 + mrow) * LDAS;
#pragma unroll
      for (int i = 0; i < 8; i++) {       // two 16B runs -> ds_load_b128
        afrag[mt][i]     = sA[row + koff + i];
        afrag[mt][8 + i] = sA[row + koff + 8 + i];
      }
    }
#pragma unroll
    for (int nt = 0; nt < 4; nt++) {
      v16bf bfrag;
      const int col = wn * 64 + nt * 16 + ncol;
#pragma unroll
      for (int e = 0; e < 16; e++)        // contiguous 32B -> ds_load_b128 x2
        bfrag[e] = sB[col * LDBS + kbase + e];
#pragma unroll
      for (int mt = 0; mt < 2; mt++)
        acc[mt][nt] = __builtin_amdgcn_wmma_f32_16x16x32_bf16(
            false, afrag[mt], false, bfrag, (short)0, acc[mt][nt], false, false);
    }
    __syncthreads();
  }

  // ---- epilogue
  const int ncol = lane & 15;
  const int mb   = (lane >> 4) * 8;
#pragma unroll
  for (int mt = 0; mt < 2; mt++)
#pragma unroll
    for (int nt = 0; nt < 4; nt++) {
      const int col = n0 + wn * 64 + nt * 16 + ncol;
      if (col >= N) continue;
#pragma unroll
      for (int r = 0; r < 8; r++) {
        const int row = m0 + wm * 32 + mt * 16 + mb + r;
        if (row >= M) continue;
        float v = acc[mt][nt][r];
        if (bias) v += bias[col];
        const size_t idx = (size_t)row * N + col;
        if (resid) v += resid[idx];
        if (C) C[idx] = v;
        if (Cbf) Cbf[idx] = (__bf16)v;
        if (Ct) Ct[((size_t)(row >> 10) * N + col) * 1024 + (row & 1023)] = (__bf16)v;
      }
    }
}

// ---------------------------------------------------------------------------
// Fused embedding + encoder LN + thalamic-gate dot: one block per token row.
// ---------------------------------------------------------------------------
__global__ __launch_bounds__(256) void embed_ln_gate_kernel(
    const int* __restrict__ tokens, const float* __restrict__ emb,
    const float* __restrict__ pos, const float* __restrict__ enw,
    const float* __restrict__ enb, const float* __restrict__ tgw,
    const float* __restrict__ tgb, float* __restrict__ xn,
    __bf16* __restrict__ xnbf, float* __restrict__ gate) {
  __shared__ float r1[8], r2[8], stats[2];
  const int row = blockIdx.x;  // 0..2047 (b*1024 + i)
  const int i   = row & 1023;
  const int tid = threadIdx.x;
  const int tok = tokens[row];
  const float* erow = emb + (size_t)tok * 2048;
  const float* prow = pos + (size_t)i * 2048;

  float vals[8], s = 0.f, ss = 0.f;
#pragma unroll
  for (int t = 0; t < 8; t++) {
    const int d = tid + t * 256;
    const float v = erow[d] + prow[d];
    vals[t] = v; s += v; ss += v * v;
  }
  for (int o = 16; o >= 1; o >>= 1) {
    s += __shfl_xor(s, o, 32);
    ss += __shfl_xor(ss, o, 32);
  }
  if ((tid & 31) == 0) { r1[tid >> 5] = s; r2[tid >> 5] = ss; }
  __syncthreads();
  if (tid == 0) {
    float a = 0.f, b = 0.f;
    for (int w = 0; w < 8; w++) { a += r1[w]; b += r2[w]; }
    const float mean = a / 2048.f;
    stats[0] = mean;
    stats[1] = rsqrtf(b / 2048.f - mean * mean + 1e-5f);
  }
  __syncthreads();
  const float mean = stats[0], inv = stats[1];
  float dot = 0.f;
#pragma unroll
  for (int t = 0; t < 8; t++) {
    const int d = tid + t * 256;
    const float v = (vals[t] - mean) * inv * enw[d] + enb[d];
    xn[(size_t)row * 2048 + d]   = v;
    xnbf[(size_t)row * 2048 + d] = (__bf16)v;
    dot += v * tgw[d];
  }
  for (int o = 16; o >= 1; o >>= 1) dot += __shfl_xor(dot, o, 32);
  if ((tid & 31) == 0) r1[tid >> 5] = dot;
  __syncthreads();
  if (tid == 0) {
    float a = 0.f;
    for (int w = 0; w < 8; w++) a += r1[w];
    gate[row] = 1.f / (1.f + __expf(-(a + tgb[0])));
  }
}

// ---------------------------------------------------------------------------
// LayerNorm, one block per row, bf16 and/or f32 outputs.
// ---------------------------------------------------------------------------
__global__ __launch_bounds__(256) void ln_kernel(
    const float* __restrict__ x, const float* __restrict__ w,
    const float* __restrict__ bv, __bf16* __restrict__ obf,
    float* __restrict__ of, int D) {
  __shared__ float r1[8], r2[8], stats[2];
  const size_t base = (size_t)blockIdx.x * D;
  const int tid = threadIdx.x;
  float s = 0.f, ss = 0.f;
  for (int d = tid; d < D; d += 256) {
    const float v = x[base + d]; s += v; ss += v * v;
  }
  for (int o = 16; o >= 1; o >>= 1) {
    s += __shfl_xor(s, o, 32);
    ss += __shfl_xor(ss, o, 32);
  }
  if ((tid & 31) == 0) { r1[tid >> 5] = s; r2[tid >> 5] = ss; }
  __syncthreads();
  if (tid == 0) {
    float a = 0.f, b = 0.f;
    for (int k = 0; k < 8; k++) { a += r1[k]; b += r2[k]; }
    const float mean = a / D;
    stats[0] = mean;
    stats[1] = rsqrtf(b / D - mean * mean + 1e-5f);
  }
  __syncthreads();
  const float mean = stats[0], inv = stats[1];
  for (int d = tid; d < D; d += 256) {
    const float v = (x[base + d] - mean) * inv * w[d] + bv[d];
    if (obf) obf[base + d] = (__bf16)v;
    if (of)  of[base + d]  = v;
  }
}

// x = xn + g[row] * y   (y already includes th_b from the GEMM epilogue)
__global__ void gate_combine_kernel(const float* __restrict__ xn,
                                    const float* __restrict__ y,
                                    const float* __restrict__ g,
                                    float* __restrict__ x, int n) {
  const int i = blockIdx.x * 256 + threadIdx.x;
  if (i < n) x[i] = xn[i] + g[i >> 11] * y[i];
}

// ff = up * silu(gate)  -> bf16 (A operand of the down-projection GEMM)
__global__ void silu_mul_kernel(const float* __restrict__ up,
                                const float* __restrict__ gt,
                                __bf16* __restrict__ ffbf, int n) {
  const int i = blockIdx.x * 256 + threadIdx.x;
  if (i < n) {
    const float g = gt[i];
    ffbf[i] = (__bf16)(up[i] * g / (1.f + __expf(-g)));
  }
}

// ---------------------------------------------------------------------------
// Fused sparse attention: one wave per (b, head, 16-query tile).
//   scores = (Q K^T) * HEAD^-0.5  via WMMA (HEAD=64 => 2 k-steps per tile)
//   per-row top-K=64 threshold via binary search on wave-reduced counts
//   softmax over kept entries, P·V via WMMA using transposed V layout.
// qbf/kbf: [b][i][h*64+d] bf16 row-major.  vT: [b][h*64+d][i] bf16.
// ---------------------------------------------------------------------------
__global__ __launch_bounds__(32) void attn_kernel(
    const __bf16* __restrict__ qbf, const __bf16* __restrict__ kbf,
    const __bf16* __restrict__ vT, __bf16* __restrict__ attnbf) {
  extern __shared__ char smem[];
  float*  sc = (float*)smem;                     // [16][1024] scores
  __bf16* pb = (__bf16*)(smem + 16 * 1024 * 4);  // [16][1024] probs

  const int blk  = blockIdx.x;
  const int qt   = blk & 63;
  const int head = (blk >> 6) & 31;
  const int b    = blk >> 11;
  const int lane = threadIdx.x;
  const int m    = lane & 15;
  const int hl   = lane >> 4;

  // Q fragments for the 16 query rows (2 chunks of d=32)
  v16bf aq[2];
  const size_t qrow = ((size_t)(b * 1024 + qt * 16 + m)) * 2048 + head * 64;
#pragma unroll
  for (int c = 0; c < 2; c++) {
    const int d0 = c * 32 + hl * 16;
#pragma unroll
    for (int i = 0; i < 8; i++) {
      aq[c][i]     = qbf[qrow + d0 + i];
      aq[c][8 + i] = qbf[qrow + d0 + 8 + i];
    }
  }

  // ---- scores: 64 key tiles of 16
  for (int jt = 0; jt < 64; jt++) {
    v8f acc = {};
#pragma unroll
    for (int c = 0; c < 2; c++) {
      v16bf bk;
      const size_t krow =
          ((size_t)(b * 1024 + jt * 16 + (lane & 15))) * 2048 + head * 64 +
          c * 32 + hl * 16;
#pragma unroll
      for (int e = 0; e < 16; e++) bk[e] = kbf[krow + e];
      acc = __builtin_amdgcn_wmma_f32_16x16x32_bf16(false, aq[c], false, bk,
                                                    (short)0, acc, false, false);
    }
    const int nn = lane & 15, mbs = hl * 8;
#pragma unroll
    for (int r = 0; r < 8; r++)
      sc[(mbs + r) * 1024 + jt * 16 + nn] = acc[r] * 0.125f;  // HEAD^-0.5
  }
  __syncthreads();

  // ---- per-row top-64 threshold + softmax
  for (int r = 0; r < 16; r++) {
    float mx = -1e30f;
    for (int t = 0; t < 32; t++) mx = fmaxf(mx, sc[r * 1024 + t * 32 + lane]);
    for (int o = 16; o >= 1; o >>= 1) mx = fmaxf(mx, __shfl_xor(mx, o, 32));
    float lo = mx - 64.f, hi = mx;
    for (int it = 0; it < 24; it++) {
      const float mid = 0.5f * (lo + hi);
      int cnt = 0;
      for (int t = 0; t < 32; t++) cnt += (sc[r * 1024 + t * 32 + lane] >= mid);
      for (int o = 16; o >= 1; o >>= 1) cnt += __shfl_xor(cnt, o, 32);
      if (cnt > 64) lo = mid; else hi = mid;
    }
    const float thr = hi;
    float sum = 0.f;
    for (int t = 0; t < 32; t++) {
      const float s = sc[r * 1024 + t * 32 + lane];
      if (s >= thr) sum += __expf(s - mx);
    }
    for (int o = 16; o >= 1; o >>= 1) sum += __shfl_xor(sum, o, 32);
    const float inv = 1.f / fmaxf(sum, 1e-20f);
    for (int t = 0; t < 32; t++) {
      const int j = t * 32 + lane;
      const float s = sc[r * 1024 + j];
      pb[r * 1024 + j] =
          (s >= thr) ? (__bf16)(__expf(s - mx) * inv) : (__bf16)0.f;
    }
  }
  __syncthreads();

  // ---- out = P @ V
  v8f oacc[4];
#pragma unroll
  for (int dt = 0; dt < 4; dt++) oacc[dt] = {};
  for (int jc = 0; jc < 32; jc++) {
    v16bf ap;
    const int koff = hl * 16;
#pragma unroll
    for (int i = 0; i < 8; i++) {
      ap[i]     = pb[m * 1024 + jc * 32 + koff + i];
      ap[8 + i] = pb[m * 1024 + jc * 32 + koff + 8 + i];
    }
#pragma unroll
    for (int dt = 0; dt < 4; dt++) {
      v16bf bv;
      const size_t vrow =
          ((size_t)b * 2048 + head * 64 + dt * 16 + (lane & 15)) * 1024 +
          jc * 32 + hl * 16;
#pragma unroll
      for (int e = 0; e < 16; e++) bv[e] = vT[vrow + e];
      oacc[dt] = __builtin_amdgcn_wmma_f32_16x16x32_bf16(
          false, ap, false, bv, (short)0, oacc[dt], false, false);
    }
  }
  const int nn = lane & 15, mbs = hl * 8;
#pragma unroll
  for (int dt = 0; dt < 4; dt++)
#pragma unroll
    for (int r = 0; r < 8; r++) {
      const size_t row = (size_t)(b * 1024 + qt * 16 + mbs + r);
      attnbf[row * 2048 + head * 64 + dt * 16 + nn] = (__bf16)oacc[dt][r];
    }
}

// ---------------------------------------------------------------------------
// Host orchestration
// ---------------------------------------------------------------------------
static inline void launch_gemm(bool bt, const __bf16* A, const float* B,
                               const float* bias, const float* resid, float* C,
                               __bf16* Cbf, __bf16* Ct, int M, int N, int K,
                               hipStream_t s) {
  dim3 g((N + BN - 1) / BN, (M + BM - 1) / BM), b(256);
  if (bt)
    gemm_kernel<true><<<g, b, 0, s>>>(A, B, bias, resid, C, Cbf, Ct, M, N, K);
  else
    gemm_kernel<false><<<g, b, 0, s>>>(A, B, bias, resid, C, Cbf, Ct, M, N, K);
}

extern "C" void kernel_launch(void* const* d_in, const int* in_sizes, int n_in,
                              void* d_out, int out_size, void* d_ws,
                              size_t ws_size, hipStream_t stream) {
  (void)in_sizes; (void)n_in; (void)out_size; (void)ws_size;
  const int M = 2048, D = 2048, HD = 2048, FF = 8192, V = 50000, L = 2;

  const int*   tokens = (const int*)d_in[0];
  const float* emb    = (const float*)d_in[1];
  const float* pos    = (const float*)d_in[2];
  const float* en_w   = (const float*)d_in[3];
  const float* en_b   = (const float*)d_in[4];
  const float* th_w   = (const float*)d_in[5];
  const float* th_b   = (const float*)d_in[6];
  const float* tg_w   = (const float*)d_in[7];
  const float* tg_b   = (const float*)d_in[8];
  const float* ln1_w  = (const float*)d_in[9];
  const float* ln1_b  = (const float*)d_in[10];
  const float* inp_w  = (const float*)d_in[11];
  const float* inp_b  = (const float*)d_in[12];
  const float* q_w    = (const float*)d_in[13];
  const float* k_w    = (const float*)d_in[14];
  const float* v_w    = (const float*)d_in[15];
  const float* o_w    = (const float*)d_in[16];
  const float* o_b    = (const float*)d_in[17];
  const float* ln2_w  = (const float*)d_in[18];
  const float* ln2_b  = (const float*)d_in[19];
  const float* up_w   = (const float*)d_in[20];
  const float* up_b   = (const float*)d_in[21];
  const float* g_w    = (const float*)d_in[22];
  const float* g_b    = (const float*)d_in[23];
  const float* dn_w   = (const float*)d_in[24];
  const float* dn_b   = (const float*)d_in[25];
  const float* on_w   = (const float*)d_in[26];
  const float* on_b   = (const float*)d_in[27];

  char* ws = (char*)d_ws;
  size_t off = 0;
  auto carve = [&](size_t bytes) -> char* {
    char* p = ws + off;
    off += (bytes + 255) & ~(size_t)255;
    return p;
  };
  float*  xn32   = (float*)carve((size_t)M * D * 4);
  float*  ybuf   = (float*)carve((size_t)M * D * 4);
  float*  gbuf   = (float*)carve((size_t)M * 4);
  float*  xbuf   = (float*)carve((size_t)M * D * 4);
  float*  hbuf   = (float*)carve((size_t)M * HD * 4);
  __bf16* lnbf   = (__bf16*)carve((size_t)M * D * 2);
  __bf16* xnbf   = (__bf16*)carve((size_t)M * D * 2);
  __bf16* hbf    = (__bf16*)carve((size_t)M * HD * 2);
  __bf16* qbf    = (__bf16*)carve((size_t)M * HD * 2);
  __bf16* kbf    = (__bf16*)carve((size_t)M * HD * 2);
  __bf16* vTbf   = (__bf16*)carve((size_t)M * HD * 2);
  __bf16* attnbf = (__bf16*)carve((size_t)M * HD * 2);
  float*  upbuf  = (float*)carve((size_t)M * FF * 4);
  float*  gtbuf  = (float*)carve((size_t)M * FF * 4);
  __bf16* ffbf   = (__bf16*)carve((size_t)M * FF * 2);

  // ---- embedding + encoder LN + gate dot
  embed_ln_gate_kernel<<<M, 256, 0, stream>>>(tokens, emb, pos, en_w, en_b,
                                              tg_w, tg_b, xn32, xnbf, gbuf);
  // y = xn @ th_w + th_b
  launch_gemm(false, xnbf, th_w, th_b, nullptr, ybuf, nullptr, nullptr, M, D, D,
              stream);
  // x = xn + g * y
  gate_combine_kernel<<<(M * D + 255) / 256, 256, 0, stream>>>(xn32, ybuf, gbuf,
                                                               xbuf, M * D);

  for (int l = 0; l < L; l++) {
    // h = ln1(x) @ inp_w + inp_b
    ln_kernel<<<M, 256, 0, stream>>>(xbuf, ln1_w + (size_t)l * D,
                                     ln1_b + (size_t)l * D, lnbf, nullptr, D);
    launch_gemm(false, lnbf, inp_w + (size_t)l * D * HD, inp_b + (size_t)l * HD,
                nullptr, hbuf, hbf, nullptr, M, HD, D, stream);
    // q, k (bf16 row-major), v (bf16 transposed for P.V fragments)
    launch_gemm(false, hbf, q_w + (size_t)l * HD * HD, nullptr, nullptr,
                nullptr, qbf, nullptr, M, HD, HD, stream);
    launch_gemm(false, hbf, k_w + (size_t)l * HD * HD, nullptr, nullptr,
                nullptr, kbf, nullptr, M, HD, HD, stream);
    launch_gemm(false, hbf, v_w + (size_t)l * HD * HD, nullptr, nullptr,
                nullptr, nullptr, vTbf, M, HD, HD, stream);
    // fused top-64 attention
    attn_kernel<<<2 * 32 * 64, 32, 98304, stream>>>(qbf, kbf, vTbf, attnbf);
    // h = h + attn @ o_w + o_b
    launch_gemm(false, attnbf, o_w + (size_t)l * HD * HD,
                o_b + (size_t)l * HD, hbuf, hbuf, nullptr, nullptr, M, HD, HD,
                stream);
    // FF: fi = ln2(h); up, gate; ff = up * silu(gate); x = h + ff @ dn_w + dn_b
    ln_kernel<<<M, 256, 0, stream>>>(hbuf, ln2_w + (size_t)l * HD,
                                     ln2_b + (size_t)l * HD, lnbf, nullptr, HD);
    launch_gemm(false, lnbf, up_w + (size_t)l * HD * FF, up_b + (size_t)l * FF,
                nullptr, upbuf, nullptr, nullptr, M, FF, HD, stream);
    launch_gemm(false, lnbf, g_w + (size_t)l * HD * FF, g_b + (size_t)l * FF,
                nullptr, gtbuf, nullptr, nullptr, M, FF, HD, stream);
    silu_mul_kernel<<<(M * FF + 255) / 256, 256, 0, stream>>>(upbuf, gtbuf,
                                                              ffbf, M * FF);
    launch_gemm(false, ffbf, dn_w + (size_t)l * FF * D, dn_b + (size_t)l * D,
                hbuf, xbuf, nullptr, nullptr, M, D, FF, stream);
  }

  // logits = ln(x) @ emb^T   (B stored [V, D] row-major -> BT GEMM)
  ln_kernel<<<M, 256, 0, stream>>>(xbuf, on_w, on_b, lnbf, nullptr, D);
  launch_gemm(true, lnbf, emb, nullptr, nullptr, (float*)d_out, nullptr,
              nullptr, M, V, D, stream);
}
